// SpikeGPT_7181185319484
// MI455X (gfx1250) — compile-verified, matrix operand order checked by reading, not run on hardware
//
#include <hip/hip_runtime.h>

typedef __attribute__((ext_vector_type(16))) _Float16 v16h;
typedef __attribute__((ext_vector_type(8)))  _Float16 v8h;
typedef __attribute__((ext_vector_type(8)))  float    v8f;

#define TT    63
#define SEQ   256
#define CH    512
#define VOC   4096
#define HEADS 8
#define HDIM  64
#define NIN   1408
#define NLAY  2
#define OD    1536              /* (H + 2G) * HD */
#define MROWS (TT * SEQ)        /* 16128 */

// ---- CDNA5 async global->LDS path (probe with __has_builtin, sync fallback)
#if defined(__gfx1250__) && __has_builtin(__builtin_amdgcn_global_load_async_to_lds_b128)
#define HAVE_ASYNC_LDS 1
typedef __attribute__((__vector_size__(4 * sizeof(int)))) int v4i;
typedef __attribute__((address_space(1))) v4i gv4i;
typedef __attribute__((address_space(3))) v4i lv4i;
#define ASYNC_LDS_B128(gp, lp) \
  __builtin_amdgcn_global_load_async_to_lds_b128((gv4i*)(gp), (lv4i*)(lp), 0, 0)
static __device__ __forceinline__ void wait_async_lds() {
#if __has_builtin(__builtin_amdgcn_s_wait_asynccnt)
  __builtin_amdgcn_s_wait_asynccnt(0);
#else
  asm volatile("s_wait_asynccnt 0x0" ::: "memory");
#endif
}
#else
#define HAVE_ASYNC_LDS 0
#endif

static __device__ __forceinline__ int ifn_count(float m0, float alpha) {
  int n = (int)ceilf(m0 / alpha) - 1;
  return n < 0 ? 0 : (n > TT ? TT : n);
}

// ---------------------------------------------------------------- elementwise
__global__ void cvt_f16_kernel(const float* __restrict__ s, _Float16* __restrict__ d, int n) {
  int i = blockIdx.x * 256 + threadIdx.x;
  if (i < n) d[i] = (_Float16)s[i];
}

__global__ void embed_kernel(const int* __restrict__ idx, const float* __restrict__ wte,
                             const float* __restrict__ poe, float* __restrict__ x) {
  int i = blockIdx.x * 256 + threadIdx.x;            // over MROWS*CH
  if (i >= MROWS * CH) return;
  int c = i & (CH - 1);
  int ts = i >> 9;                                   // t*SEQ + s
  int s = ts & (SEQ - 1);
  x[i] = wte[(size_t)idx[ts] * CH + c] + poe[(s << 9) + c];
}

// rstd per (t,s) row: rsqrt(mean(x^2)+eps)
__global__ __launch_bounds__(256) void rstd_kernel(const float* __restrict__ x,
                                                   float* __restrict__ rstd) {
  __shared__ float red[256];
  int row = blockIdx.x;
  const float* xr = x + (size_t)row * CH;
  float s = 0.f;
  for (int c = threadIdx.x; c < CH; c += 256) { float v = xr[c]; s += v * v; }
  red[threadIdx.x] = s; __syncthreads();
  for (int off = 128; off > 0; off >>= 1) {
    if (threadIdx.x < off) red[threadIdx.x] += red[threadIdx.x + off];
    __syncthreads();
  }
  if (threadIdx.x == 0) rstd[row] = rsqrtf(red[0] * (1.f / CH) + 1e-5f);
}

// if_neuron counts of rmsnorm'd x, reduced over T per (s,c)
__global__ void ifn_counts_kernel(const float* __restrict__ x, const float* __restrict__ rstd,
                                  const float* __restrict__ nw, const float* __restrict__ alpha_p,
                                  int* __restrict__ cnt) {
  int i = blockIdx.x * 256 + threadIdx.x;            // over SEQ*CH
  if (i >= SEQ * CH) return;
  int s = i >> 9, c = i & (CH - 1);
  float sum = 0.f;
  for (int t = 0; t < TT; ++t)
    sum += x[((size_t)t * SEQ + s) * CH + c] * rstd[t * SEQ + s];
  sum *= nw[c];
  float a = *alpha_p;
  cnt[i] = ifn_count(0.5f * a + sum, a);
}

__global__ void scale_cnt_kernel(const int* __restrict__ cnt, const float* __restrict__ alpha_p,
                                 _Float16* __restrict__ d) {
  int i = blockIdx.x * 256 + threadIdx.x;
  if (i >= SEQ * CH) return;
  d[i] = (_Float16)(*alpha_p * (float)cnt[i]);       // = sum over t of spikes (exact in f16)
}

// qkv_sum -> if_neuron count -> mean over T, scatter into qm/km/vm [h][s][d]
__global__ void qkv_stats_kernel(const float* __restrict__ qkvsum, const float* __restrict__ alpha_p,
                                 float* __restrict__ qm, float* __restrict__ km,
                                 float* __restrict__ vm) {
  int i = blockIdx.x * 256 + threadIdx.x;            // over SEQ*OD
  if (i >= SEQ * OD) return;
  int s = i / OD, o = i - s * OD;
  float a = *alpha_p;
  int n = ifn_count(0.5f * a + qkvsum[i], a);
  float mean = a * (float)n * (1.f / TT);
  int g = o / 192, j = (o / 64) % 3, d = o & 63;
  float* dst = (j == 0) ? qm : (j == 1) ? km : vm;
  dst[((size_t)g * SEQ + s) * HDIM + d] = mean;
}

// causal scores + softmax + spiked-probability mean; one block per (h,q)
__global__ __launch_bounds__(256) void attn_kernel(const float* __restrict__ qm,
                                                   const float* __restrict__ km,
                                                   const float* __restrict__ alpha_p,
                                                   float* __restrict__ psm) {
  __shared__ float qrow[HDIM];
  __shared__ float red[256];
  int q = blockIdx.x, h = blockIdx.y, k = threadIdx.x;
  if (k < HDIM) qrow[k] = qm[((size_t)h * SEQ + q) * HDIM + k];
  __syncthreads();
  float val = -3.4e38f;
  if (k <= q) {
    const float* kr = km + ((size_t)h * SEQ + k) * HDIM;
    float acc = 0.f;
#pragma unroll
    for (int d = 0; d < HDIM; ++d) acc += qrow[d] * kr[d];
    val = acc * 0.125f;                              // 1/sqrt(64)
  }
  red[k] = val; __syncthreads();
  for (int off = 128; off > 0; off >>= 1) {
    if (k < off) red[k] = fmaxf(red[k], red[k + off]);
    __syncthreads();
  }
  float mx = red[0]; __syncthreads();
  float e = (k <= q) ? __expf(val - mx) : 0.f;
  red[k] = e; __syncthreads();
  for (int off = 128; off > 0; off >>= 1) {
    if (k < off) red[k] += red[k + off];
    __syncthreads();
  }
  float p = e / red[0];
  float a = *alpha_p;
  int n = ifn_count(0.5f * a + (float)TT * p, a);    // p broadcast over T
  psm[((size_t)h * SEQ + q) * SEQ + k] = a * (float)n * (1.f / TT);
}

// y[h][q][d] = sum_k psm * vm ; pack into yflat16[s][h*64+d]
__global__ void attn_y_kernel(const float* __restrict__ psm, const float* __restrict__ vm,
                              _Float16* __restrict__ yflat) {
  int i = blockIdx.x * 256 + threadIdx.x;            // HEADS*SEQ*HDIM = 131072
  if (i >= HEADS * SEQ * HDIM) return;
  int d = i & 63, q = (i >> 6) & (SEQ - 1), h = i >> 14;
  const float* pr = psm + ((size_t)h * SEQ + q) * SEQ;
  float acc = 0.f;
  for (int k = 0; k < SEQ; ++k) acc += pr[k] * vm[((size_t)h * SEQ + k) * HDIM + d];
  yflat[(size_t)q * CH + h * HDIM + d] = (_Float16)acc;
}

// x[t,s,c] += attnout[s,c]  (attention contribution is T-invariant)
__global__ void add_bcast_kernel(float* __restrict__ x, const float* __restrict__ ao) {
  int i = blockIdx.x * 256 + threadIdx.x;
  if (i >= MROWS * CH) return;
  x[i] += ao[i % (SEQ * CH)];
}

// expand counts -> f16 spike tensor (t,s,c)
__global__ void spikes_kernel(const int* __restrict__ cnt, const float* __restrict__ alpha_p,
                              _Float16* __restrict__ spk) {
  int i = blockIdx.x * 256 + threadIdx.x;
  if (i >= MROWS * CH) return;
  int sc = i % (SEQ * CH);
  int t = i / (SEQ * CH);
  spk[i] = (_Float16)((t < cnt[sc]) ? *alpha_p : 0.f);
}

// ------------------------------------------------------------------ WMMA GEMM
// C[M,N] = A[M,K] * B[N,K]^T ; MODE 0: store f32, 1: += f32 (residual),
// 2: dual-B, Hout = silu(A*B1^T) * (A*B2^T) in f16.
// Block: 256 thr = 8 waves; wave computes MT m-tiles (rows +0, +128) x 64 N.
// B tiles double-buffered in LDS, staged with async global->LDS when available.
template <int MODE, int MT>
__global__ __launch_bounds__(256) void gemm16_kernel(const _Float16* __restrict__ A,
                                                     const _Float16* __restrict__ B,
                                                     const _Float16* __restrict__ B2,
                                                     float* __restrict__ C,
                                                     _Float16* __restrict__ Hout,
                                                     int N, int K) {
  __shared__ __align__(16) _Float16 ldsB[2][64][40];
  __shared__ __align__(16) _Float16 ldsB2[(MODE == 2) ? 2 : 1][64][40];
  const int tid = threadIdx.x;
  const int wave = tid >> 5, lane = tid & 31;
  const int lr = lane & 15;                // row-in-tile (A) / col-in-tile (B)
  const int kl = (lane >> 4) << 3;         // 0 or 8: K sub-block per half-wave
  const int nBase = blockIdx.x * 64;
  const int mBase = blockIdx.y * (128 * MT) + wave * 16;

  v8f acc[MT][4];
  v8f acc2[4];
#pragma unroll
  for (int mt = 0; mt < MT; ++mt)
#pragma unroll
    for (int nt = 0; nt < 4; ++nt) acc[mt][nt] = v8f{};
#pragma unroll
  for (int nt = 0; nt < 4; ++nt) acc2[nt] = v8f{};

  const _Float16* Arow[MT];
#pragma unroll
  for (int mt = 0; mt < MT; ++mt)
    Arow[mt] = A + (size_t)(mBase + mt * 128 + lr) * K;

  const int ldrow = tid >> 2;              // 0..63
  const int ldseg = (tid & 3) << 3;        // 0,8,16,24
  const _Float16* Bld = B + (size_t)(nBase + ldrow) * K + ldseg;
  const _Float16* B2ld = (MODE == 2) ? (B2 + (size_t)(nBase + ldrow) * K + ldseg) : nullptr;

  auto stage = [&](int kk, int buf) {
#if HAVE_ASYNC_LDS
    ASYNC_LDS_B128(Bld + kk, &ldsB[buf][ldrow][ldseg]);
    if constexpr (MODE == 2) ASYNC_LDS_B128(B2ld + kk, &ldsB2[buf][ldrow][ldseg]);
#else
    *(v8h*)&ldsB[buf][ldrow][ldseg] = *(const v8h*)(Bld + kk);
    if constexpr (MODE == 2) *(v8h*)&ldsB2[buf][ldrow][ldseg] = *(const v8h*)(B2ld + kk);
#endif
  };

  stage(0, 0);                             // prologue
  for (int kk = 0; kk < K; kk += 32) {
    const int cur = (kk >> 5) & 1;
#if HAVE_ASYNC_LDS
    wait_async_lds();                      // own async copies landed in LDS
#endif
    __syncthreads();                       // all waves' copies visible; prev reads done
    if (kk + 32 < K) {
      stage(kk + 32, cur ^ 1);             // overlap next B tile with compute
      __builtin_prefetch(Arow[0] + kk + 32, 0, 1);   // global_prefetch_b8
    }

    v16h a[MT];
#pragma unroll
    for (int mt = 0; mt < MT; ++mt) {
      v8h lo = *(const v8h*)(Arow[mt] + kk + kl);      // K = kl+0..7
      v8h hi = *(const v8h*)(Arow[mt] + kk + kl + 16); // K = kl+16..23
#pragma unroll
      for (int i = 0; i < 8; ++i) { a[mt][i] = lo[i]; a[mt][i + 8] = hi[i]; }
    }
#pragma unroll
    for (int nt = 0; nt < 4; ++nt) {
      v16h b;
      {
        v8h lo = *(const v8h*)&ldsB[cur][nt * 16 + lr][kl];
        v8h hi = *(const v8h*)&ldsB[cur][nt * 16 + lr][kl + 16];
#pragma unroll
        for (int i = 0; i < 8; ++i) { b[i] = lo[i]; b[i + 8] = hi[i]; }
      }
#pragma unroll
      for (int mt = 0; mt < MT; ++mt)
        acc[mt][nt] = __builtin_amdgcn_wmma_f32_16x16x32_f16(false, a[mt], false, b, (short)0,
                                                             acc[mt][nt], false, false);
      if constexpr (MODE == 2) {
        v16h b2;
        v8h lo = *(const v8h*)&ldsB2[cur][nt * 16 + lr][kl];
        v8h hi = *(const v8h*)&ldsB2[cur][nt * 16 + lr][kl + 16];
#pragma unroll
        for (int i = 0; i < 8; ++i) { b2[i] = lo[i]; b2[i + 8] = hi[i]; }
        acc2[nt] = __builtin_amdgcn_wmma_f32_16x16x32_f16(false, a[0], false, b2, (short)0,
                                                          acc2[nt], false, false);
      }
    }
  }

#pragma unroll
  for (int mt = 0; mt < MT; ++mt) {
    const int mOut = mBase + mt * 128 + ((lane >> 4) << 3);  // half-wave covers M rows 8..15
#pragma unroll
    for (int nt = 0; nt < 4; ++nt) {
      const int n = nBase + nt * 16 + lr;
#pragma unroll
      for (int r = 0; r < 8; ++r) {
        const size_t idx = (size_t)(mOut + r) * N + n;
        if constexpr (MODE == 0) {
          C[idx] = acc[mt][nt][r];
        } else if constexpr (MODE == 1) {
          C[idx] += acc[mt][nt][r];
        } else {
          float g = acc[mt][nt][r];
          float h = g * (1.f / (1.f + __expf(-g))) * acc2[nt][r];  // silu(g1)*g2
          Hout[idx] = (_Float16)h;
        }
      }
    }
  }
}

// ------------------------------------------------------------------ host side
extern "C" void kernel_launch(void* const* d_in, const int* in_sizes, int n_in,
                              void* d_out, int out_size, void* d_ws, size_t ws_size,
                              hipStream_t stream) {
  (void)in_sizes; (void)n_in; (void)out_size; (void)ws_size;
  const int*   idx       = (const int*)d_in[0];
  const float* wte       = (const float*)d_in[1];
  const float* poe       = (const float*)d_in[2];
  const float* ln_f_w    = (const float*)d_in[3];
  const float* lm_head_w = (const float*)d_in[4];
  const float* out_alpha = (const float*)d_in[5];
  const float* norm1_w   = (const float*)d_in[6];
  const float* norm2_w   = (const float*)d_in[7];
  const float* attn_w    = (const float*)d_in[8];
  const float* proj_w    = (const float*)d_in[9];
  const float* w1        = (const float*)d_in[10];
  const float* w2        = (const float*)d_in[11];
  const float* w3        = (const float*)d_in[12];
  const float* alphas    = (const float*)d_in[13];
  float* out = (float*)d_out;

  char* base = (char*)d_ws;
  size_t off = 0;
  auto alloc = [&](size_t bytes) -> char* {
    char* p = base + off;
    off += (bytes + 255) & ~(size_t)255;
    return p;
  };

  float*    xf      = (float*)alloc((size_t)MROWS * CH * 4);
  float*    rstd    = (float*)alloc((size_t)MROWS * 4);
  int*      cnt     = (int*)alloc((size_t)SEQ * CH * 4);
  _Float16* acnt    = (_Float16*)alloc((size_t)SEQ * CH * 2);
  float*    qkvsum  = (float*)alloc((size_t)SEQ * OD * 4);
  float*    qm      = (float*)alloc((size_t)HEADS * SEQ * HDIM * 4);
  float*    km      = (float*)alloc((size_t)HEADS * SEQ * HDIM * 4);
  float*    vm      = (float*)alloc((size_t)HEADS * SEQ * HDIM * 4);
  float*    psm     = (float*)alloc((size_t)HEADS * SEQ * SEQ * 4);
  _Float16* yflat   = (_Float16*)alloc((size_t)SEQ * CH * 2);
  float*    attnout = (float*)alloc((size_t)SEQ * CH * 4);
  _Float16* spk     = (_Float16*)alloc((size_t)MROWS * CH * 2);
  _Float16* hbuf    = (_Float16*)alloc((size_t)MROWS * NIN * 2);
  _Float16* attn16  = (_Float16*)alloc((size_t)NLAY * OD * CH * 2);
  _Float16* proj16  = (_Float16*)alloc((size_t)NLAY * CH * CH * 2);
  _Float16* w116    = (_Float16*)alloc((size_t)NLAY * NIN * CH * 2);
  _Float16* w216    = (_Float16*)alloc((size_t)NLAY * NIN * CH * 2);
  _Float16* w316    = (_Float16*)alloc((size_t)NLAY * CH * NIN * 2);
  _Float16* lm16    = (_Float16*)alloc((size_t)VOC * CH * 2);

  auto cvt = [&](const float* s, _Float16* d, size_t n) {
    cvt_f16_kernel<<<dim3((unsigned)((n + 255) / 256)), dim3(256), 0, stream>>>(s, d, (int)n);
  };
  cvt(attn_w, attn16, (size_t)NLAY * OD * CH);
  cvt(proj_w, proj16, (size_t)NLAY * CH * CH);
  cvt(w1, w116, (size_t)NLAY * NIN * CH);
  cvt(w2, w216, (size_t)NLAY * NIN * CH);
  cvt(w3, w316, (size_t)NLAY * CH * NIN);
  cvt(lm_head_w, lm16, (size_t)VOC * CH);

  const unsigned gMC = (MROWS * CH + 255) / 256;
  const unsigned gSC = (SEQ * CH + 255) / 256;

  embed_kernel<<<dim3(gMC), dim3(256), 0, stream>>>(idx, wte, poe, xf);

  for (int l = 0; l < NLAY; ++l) {
    const float* a0 = alphas + l * 4 + 0;
    const float* a1 = alphas + l * 4 + 1;
    const float* a2 = alphas + l * 4 + 2;
    const float* a3 = alphas + l * 4 + 3;

    // ---- attention branch (collapsed over T)
    rstd_kernel<<<dim3(MROWS), dim3(256), 0, stream>>>(xf, rstd);
    ifn_counts_kernel<<<dim3(gSC), dim3(256), 0, stream>>>(xf, rstd, norm1_w + l * CH, a0, cnt);
    scale_cnt_kernel<<<dim3(gSC), dim3(256), 0, stream>>>(cnt, a0, acnt);
    gemm16_kernel<0, 2><<<dim3(OD / 64, SEQ / 256), dim3(256), 0, stream>>>(
        acnt, attn16 + (size_t)l * OD * CH, nullptr, qkvsum, nullptr, OD, CH);
    qkv_stats_kernel<<<dim3((SEQ * OD + 255) / 256), dim3(256), 0, stream>>>(qkvsum, a2, qm, km, vm);
    attn_kernel<<<dim3(SEQ, HEADS), dim3(256), 0, stream>>>(qm, km, a3, psm);
    attn_y_kernel<<<dim3((HEADS * SEQ * HDIM + 255) / 256), dim3(256), 0, stream>>>(psm, vm, yflat);
    gemm16_kernel<0, 2><<<dim3(CH / 64, SEQ / 256), dim3(256), 0, stream>>>(
        yflat, proj16 + (size_t)l * CH * CH, nullptr, attnout, nullptr, CH, CH);
    add_bcast_kernel<<<dim3(gMC), dim3(256), 0, stream>>>(xf, attnout);

    // ---- MLP branch (T-dependent)
    rstd_kernel<<<dim3(MROWS), dim3(256), 0, stream>>>(xf, rstd);
    ifn_counts_kernel<<<dim3(gSC), dim3(256), 0, stream>>>(xf, rstd, norm2_w + l * CH, a1, cnt);
    spikes_kernel<<<dim3(gMC), dim3(256), 0, stream>>>(cnt, a1, spk);
    gemm16_kernel<2, 1><<<dim3(NIN / 64, MROWS / 128), dim3(256), 0, stream>>>(
        spk, w116 + (size_t)l * NIN * CH, w216 + (size_t)l * NIN * CH, nullptr, hbuf, NIN, CH);
    gemm16_kernel<1, 2><<<dim3(CH / 64, MROWS / 256), dim3(256), 0, stream>>>(
        hbuf, w316 + (size_t)l * CH * NIN, nullptr, xf, nullptr, CH, NIN);
  }

  // ---- final norm + spike + lm_head
  rstd_kernel<<<dim3(MROWS), dim3(256), 0, stream>>>(xf, rstd);
  ifn_counts_kernel<<<dim3(gSC), dim3(256), 0, stream>>>(xf, rstd, ln_f_w, out_alpha, cnt);
  spikes_kernel<<<dim3(gMC), dim3(256), 0, stream>>>(cnt, out_alpha, spk);
  gemm16_kernel<0, 2><<<dim3(VOC / 64, MROWS / 256), dim3(256), 0, stream>>>(
      spk, lm16, nullptr, out, nullptr, VOC, CH);
}
